// PDE_M2_72911364817679
// MI455X (gfx1250) — matrix operand order repaired and checked by reading, not compiled, not run on hardware
//
#include <hip/hip_runtime.h>
#include <hip/hip_bf16.h>
#include <math.h>

typedef __attribute__((ext_vector_type(16))) _Float16 v16h;
typedef __attribute__((ext_vector_type(8)))  float    v8f;

#define HIDDEN  128
#define MSG_DIM 64
#define DTC     0.01f

// tanh via hardware v_exp_f32; |x| clamped where f32 tanh saturates anyway
__device__ __forceinline__ float fast_tanh(float x) {
  x = fminf(fmaxf(x, -10.f), 10.f);
  float e = __expf(2.0f * x);
  return (e - 1.0f) * __builtin_amdgcn_rcpf(e + 1.0f);
}
__device__ __forceinline__ float fast_softplus(float x) {
  return (x > 20.f) ? x : __logf(1.0f + __expf(x));
}
__device__ __forceinline__ float fast_pow10(float x) {
  return __expf(x * 2.302585093f);
}

// ---------------------------------------------------------------- init
__global__ void k_init(float* __restrict__ h_rxn, float* __restrict__ ext_agg,
                       float* __restrict__ n_sub, unsigned* __restrict__ rxn_scale,
                       float* __restrict__ tot_cons, float* __restrict__ dxdt,
                       int n_rxn, int n_met) {
  long long i = (long long)blockIdx.x * blockDim.x + threadIdx.x;
  if (i < (long long)n_rxn * MSG_DIM) h_rxn[i] = 0.f;
  if (i < n_rxn) { ext_agg[i] = 0.f; n_sub[i] = 0.f; rxn_scale[i] = 0x3F800000u; }
  if (i < n_met) { tot_cons[i] = 0.f; dxdt[i] = 0.f; }
}

// ------------------------------------------------- edge MLP + scatter
// One wave = 16 edges. h = tanh([conc,sto]@W1+b1) built per-lane directly in
// WMMA A-fragment layout; msg = h@W2 via v_wmma_f32_16x16x32_f16.
// W2 is staged in LDS pre-permuted into fragment order so each B fragment is
// one contiguous 32-byte v16h load (2x ds_load_b128).
__global__ __launch_bounds__(256)
void k_edge_mlp(const float* __restrict__ x,
                const int*   __restrict__ met_sub,
                const int*   __restrict__ rxn_sub,
                const float* __restrict__ sto_sub,
                const float* __restrict__ W1, const float* __restrict__ b1,
                const float* __restrict__ W2, const float* __restrict__ b2,
                float* __restrict__ h_rxn, float* __restrict__ ext_agg,
                float* __restrict__ n_sub, int E) {
  __shared__ __align__(16) float    sW1[2 * HIDDEN];
  __shared__ __align__(16) float    sB1[HIDDEN];
  __shared__ __align__(32) _Float16 sW2f[4 * 4 * 32 * 16];  // [nt][kb][lane][i]
  __shared__ float                  sB2[MSG_DIM];

  for (int i = threadIdx.x; i < 2 * HIDDEN; i += blockDim.x) sW1[i] = W1[i];
  for (int i = threadIdx.x; i < HIDDEN; i += blockDim.x)     sB1[i] = b1[i];
  for (int i = threadIdx.x; i < MSG_DIM; i += blockDim.x)    sB2[i] = b2[i];
  // permute W2 (128x64 row-major) into B-fragment order:
  //   elem i of fragment (nt,kb) in lane (hi*16+col) = W2[(kb*32+(i>>3)*16+hi*8+(i&7))*64 + nt*16+col]
  for (int s = threadIdx.x; s < HIDDEN * MSG_DIM; s += blockDim.x) {
    int k  = s >> 6,  n  = s & 63;
    int nt = n >> 4,  cl = n & 15;
    int kb = k >> 5,  kl = k & 31;
    int h2 = (kl >> 3) & 1, ii = (kl & 7) | ((kl >> 4) << 3);
    sW2f[(((nt * 4 + kb) * 32) + h2 * 16 + cl) * 16 + ii] = (_Float16)W2[s];
  }
  __syncthreads();

  const int wave = threadIdx.x >> 5;
  const int lane = threadIdx.x & 31;
  const int row  = lane & 15;
  const int hi   = lane >> 4;
  const int eBase = (blockIdx.x * 8 + wave) * 16;

  int  e      = eBase + row;
  bool evalid = (e < E);
  int  eC     = evalid ? e : 0;
  float sto   = evalid ? sto_sub[eC] : 0.f;
  int   met   = met_sub[eC];
  float conc  = x[met * 8 + 3];
  float ext   = x[met * 8 + 4];

  if (hi == 0 && evalid) {                       // once per edge
    int r = rxn_sub[eC];
    atomicAdd(&ext_agg[r], ext);
    atomicAdd(&n_sub[r], 1.0f);
  }

  // A fragments (16-bit A 16x32): lane's K-set is two contiguous 8-wide groups
  v16h a[4];
#pragma unroll
  for (int kb = 0; kb < 4; ++kb) {
    const int b0 = kb * 32 + hi * 8;
    const int b1i = b0 + 16;
    float w0[16], w1v[16], bb[16];
    *(float4*)&w0[0]   = *(const float4*)&sW1[b0];
    *(float4*)&w0[4]   = *(const float4*)&sW1[b0 + 4];
    *(float4*)&w0[8]   = *(const float4*)&sW1[b1i];
    *(float4*)&w0[12]  = *(const float4*)&sW1[b1i + 4];
    *(float4*)&w1v[0]  = *(const float4*)&sW1[HIDDEN + b0];
    *(float4*)&w1v[4]  = *(const float4*)&sW1[HIDDEN + b0 + 4];
    *(float4*)&w1v[8]  = *(const float4*)&sW1[HIDDEN + b1i];
    *(float4*)&w1v[12] = *(const float4*)&sW1[HIDDEN + b1i + 4];
    *(float4*)&bb[0]   = *(const float4*)&sB1[b0];
    *(float4*)&bb[4]   = *(const float4*)&sB1[b0 + 4];
    *(float4*)&bb[8]   = *(const float4*)&sB1[b1i];
    *(float4*)&bb[12]  = *(const float4*)&sB1[b1i + 4];
#pragma unroll
    for (int i = 0; i < 16; ++i)
      a[kb][i] = (_Float16)fast_tanh(conc * w0[i] + sto * w1v[i] + bb[i]);
  }

  // reaction ids for the C-tile rows this lane owns (M = hi*8 + r)
  int  rid[8]; bool rvalid[8];
#pragma unroll
  for (int r = 0; r < 8; ++r) {
    int em    = eBase + hi * 8 + r;
    rvalid[r] = (em < E);
    rid[r]    = rxn_sub[rvalid[r] ? em : 0];
  }

  const int col = row;
#pragma unroll
  for (int nt = 0; nt < 4; ++nt) {
    v8f c = {};
#pragma unroll
    for (int kb = 0; kb < 4; ++kb) {
      v16h bf = *(const v16h*)&sW2f[(((nt * 4 + kb) * 32) + lane) * 16];
      c = __builtin_amdgcn_wmma_f32_16x16x32_f16(false, a[kb], false, bf,
                                                 (short)0, c, false, false);
    }
    float bias = sB2[nt * 16 + col];
#pragma unroll
    for (int r = 0; r < 8; ++r)
      if (rvalid[r])
        atomicAdd(&h_rxn[(long long)rid[r] * MSG_DIM + nt * 16 + col], c[r] + bias);
  }
}

// ------------------------------------------------ reaction MLP (WMMA)
__global__ __launch_bounds__(256)
void k_rxn_mlp(const float* __restrict__ h_rxn,
               const float* __restrict__ V1, const float* __restrict__ c1,
               const float* __restrict__ V2, const float* __restrict__ c2,
               const float* __restrict__ log_k,
               const float* __restrict__ ext_agg, const float* __restrict__ n_sub,
               float* __restrict__ v_pre, int n_rxn) {
  __shared__ __align__(32) _Float16 sV1f[8 * 2 * 32 * 16];  // [nt][kb][lane][i]
  __shared__ float sC1[HIDDEN];
  __shared__ float sV2[HIDDEN];
  // permute V1 (64x128 row-major) into B-fragment order
  for (int s = threadIdx.x; s < MSG_DIM * HIDDEN; s += blockDim.x) {
    int k  = s >> 7,  n  = s & 127;
    int nt = n >> 4,  cl = n & 15;
    int kb = k >> 5,  kl = k & 31;
    int h2 = (kl >> 3) & 1, ii = (kl & 7) | ((kl >> 4) << 3);
    sV1f[(((nt * 2 + kb) * 32) + h2 * 16 + cl) * 16 + ii] = (_Float16)V1[s];
  }
  for (int i = threadIdx.x; i < HIDDEN; i += blockDim.x) { sC1[i] = c1[i]; sV2[i] = V2[i]; }
  __syncthreads();

  const int wave  = threadIdx.x >> 5;
  const int lane  = threadIdx.x & 31;
  const int row   = lane & 15;
  const int hi    = lane >> 4;
  const int rBase = (blockIdx.x * 8 + wave) * 16;

  int rr = rBase + row;  if (rr >= n_rxn) rr = n_rxn - 1;   // clamp; masked on store
  const float* hrow = &h_rxn[(long long)rr * MSG_DIM];

  // A fragments: lane's K-set = two contiguous 8-float groups -> global b128 loads
  v16h a[2];
#pragma unroll
  for (int kb = 0; kb < 2; ++kb) {
    const int b0 = kb * 32 + hi * 8;
    float hv[16];
    *(float4*)&hv[0]  = *(const float4*)&hrow[b0];
    *(float4*)&hv[4]  = *(const float4*)&hrow[b0 + 4];
    *(float4*)&hv[8]  = *(const float4*)&hrow[b0 + 16];
    *(float4*)&hv[12] = *(const float4*)&hrow[b0 + 20];
#pragma unroll
    for (int i = 0; i < 16; ++i) a[kb][i] = (_Float16)hv[i];
  }

  float s[8];
#pragma unroll
  for (int r = 0; r < 8; ++r) s[r] = 0.f;

  const int col = row;
#pragma unroll
  for (int nt = 0; nt < 8; ++nt) {
    int Ng = nt * 16 + col;
    v8f c = {};
#pragma unroll
    for (int kb = 0; kb < 2; ++kb) {
      v16h bf = *(const v16h*)&sV1f[(((nt * 2 + kb) * 32) + lane) * 16];
      c = __builtin_amdgcn_wmma_f32_16x16x32_f16(false, a[kb], false, bf,
                                                 (short)0, c, false, false);
    }
    float w2v = sV2[Ng], c1v = sC1[Ng];
#pragma unroll
    for (int r = 0; r < 8; ++r)
      s[r] += fast_tanh(c[r] + c1v) * w2v;
  }

  // reduce the length-128 dot across the 16 lanes of this half-wave
#pragma unroll
  for (int r = 0; r < 8; ++r)
#pragma unroll
    for (int m = 1; m < 16; m <<= 1)
      s[r] += __shfl_xor(s[r], m, 16);

  if (col == 0) {
    float c2v = c2[0];
#pragma unroll
    for (int r = 0; r < 8; ++r) {
      int rxn = rBase + hi * 8 + r;
      if (rxn < n_rxn) {
        float ns  = n_sub[rxn]; ns = (ns < 1.f) ? 1.f : ns;
        float em  = ext_agg[rxn] / ns;
        float k10 = fast_pow10(log_k[rxn]);
        v_pre[rxn] = k10 * em * fast_softplus(s[r] + c2v);
      }
    }
  }
}

// -------------------------------------------- scalar scatter stages
__global__ void k_consumption(const int* __restrict__ met_sub,
                              const int* __restrict__ rxn_sub,
                              const float* __restrict__ sto_sub,
                              const float* __restrict__ v_pre,
                              float* __restrict__ tot_cons, int E) {
  int e = blockIdx.x * blockDim.x + threadIdx.x;
  if (e < E)
    atomicAdd(&tot_cons[met_sub[e]], sto_sub[e] * v_pre[rxn_sub[e]] * DTC);
}

__global__ void k_met_scale(const float* __restrict__ x,
                            const float* __restrict__ tot_cons,
                            float* __restrict__ met_scale, int n_met) {
  int m = blockIdx.x * blockDim.x + threadIdx.x;
  if (m < n_met) {
    float tot = tot_cons[m];
    float sc  = 1.f;
    if (tot > 1e-12f) { float q = x[m * 8 + 3] / tot; sc = (q < 1.f) ? q : 1.f; }
    met_scale[m] = sc;
  }
}

// segment-min via atomicMin on bit pattern (scales are non-negative floats)
__global__ void k_rxn_scale(const int* __restrict__ met_sub,
                            const int* __restrict__ rxn_sub,
                            const float* __restrict__ met_scale,
                            unsigned* __restrict__ rxn_scale, int E) {
  int e = blockIdx.x * blockDim.x + threadIdx.x;
  if (e < E)
    atomicMin(&rxn_scale[rxn_sub[e]], __float_as_uint(met_scale[met_sub[e]]));
}

__global__ void k_dxdt(const int* __restrict__ met_sub,  const int* __restrict__ rxn_sub,
                       const float* __restrict__ sto_sub,
                       const int* __restrict__ met_prod, const int* __restrict__ rxn_prod,
                       const float* __restrict__ sto_prod,
                       const float* __restrict__ v_pre,
                       const unsigned* __restrict__ rxn_scale,
                       float* __restrict__ dxdt, int Es, int Ep) {
  int e = blockIdx.x * blockDim.x + threadIdx.x;
  if (e < Es) {
    int r   = rxn_sub[e];
    float v = v_pre[r] * __uint_as_float(rxn_scale[r]);
    atomicAdd(&dxdt[met_sub[e]], -sto_sub[e] * v);
  } else if (e < Es + Ep) {
    int i   = e - Es;
    int r   = rxn_prod[i];
    float v = v_pre[r] * __uint_as_float(rxn_scale[r]);
    atomicAdd(&dxdt[met_prod[i]], sto_prod[i] * v);
  }
}

// ---------------------------------------------------------------- launch
extern "C" void kernel_launch(void* const* d_in, const int* in_sizes, int n_in,
                              void* d_out, int out_size, void* d_ws, size_t ws_size,
                              hipStream_t stream) {
  const float* x        = (const float*)d_in[0];
  const int*   met_sub  = (const int*)  d_in[1];
  const int*   rxn_sub  = (const int*)  d_in[2];
  const float* sto_sub  = (const float*)d_in[3];
  const int*   met_prod = (const int*)  d_in[4];
  const int*   rxn_prod = (const int*)  d_in[5];
  const float* sto_prod = (const float*)d_in[6];
  const float* W1       = (const float*)d_in[7];
  const float* b1       = (const float*)d_in[8];
  const float* W2       = (const float*)d_in[9];
  const float* b2       = (const float*)d_in[10];
  const float* V1       = (const float*)d_in[11];
  const float* c1       = (const float*)d_in[12];
  const float* V2       = (const float*)d_in[13];
  const float* c2       = (const float*)d_in[14];
  const float* log_k    = (const float*)d_in[15];

  const int n_met = in_sizes[0] / 8;
  const int Es    = in_sizes[1];
  const int Ep    = in_sizes[4];
  const int n_rxn = in_sizes[15];

  // workspace carve-up (~55 MB)
  float* ws        = (float*)d_ws;
  float* h_rxn     = ws;                         ws += (size_t)n_rxn * MSG_DIM;
  float* ext_agg   = ws;                         ws += n_rxn;
  float* n_sub     = ws;                         ws += n_rxn;
  float* v_pre     = ws;                         ws += n_rxn;
  float* tot_cons  = ws;                         ws += n_met;
  float* met_scale = ws;                         ws += n_met;
  unsigned* rxn_scale = (unsigned*)ws;
  float* dxdt      = (float*)d_out;

  long long initN = (long long)n_rxn * MSG_DIM;
  int initBlocks  = (int)((initN + 255) / 256);
  k_init<<<initBlocks, 256, 0, stream>>>(h_rxn, ext_agg, n_sub, rxn_scale,
                                         tot_cons, dxdt, n_rxn, n_met);

  k_edge_mlp<<<(Es + 127) / 128, 256, 0, stream>>>(
      x, met_sub, rxn_sub, sto_sub, W1, b1, W2, b2, h_rxn, ext_agg, n_sub, Es);

  k_rxn_mlp<<<(n_rxn + 127) / 128, 256, 0, stream>>>(
      h_rxn, V1, c1, V2, c2, log_k, ext_agg, n_sub, v_pre, n_rxn);

  k_consumption<<<(Es + 255) / 256, 256, 0, stream>>>(
      met_sub, rxn_sub, sto_sub, v_pre, tot_cons, Es);

  k_met_scale<<<(n_met + 255) / 256, 256, 0, stream>>>(x, tot_cons, met_scale, n_met);

  k_rxn_scale<<<(Es + 255) / 256, 256, 0, stream>>>(
      met_sub, rxn_sub, met_scale, rxn_scale, Es);

  k_dxdt<<<(Es + Ep + 255) / 256, 256, 0, stream>>>(
      met_sub, rxn_sub, sto_sub, met_prod, rxn_prod, sto_prod,
      v_pre, rxn_scale, dxdt, Es, Ep);
}
